// AttentionBlock_28363964023011
// MI455X (gfx1250) — matrix operand order repaired and checked by reading, not compiled, not run on hardware
//
#include <hip/hip_runtime.h>
#include <hip/hip_bf16.h>

// ---------- types ----------
typedef __attribute__((ext_vector_type(16))) __bf16        v16bf;
typedef __attribute__((ext_vector_type(8)))  float         v8f;
typedef __attribute__((ext_vector_type(16))) unsigned short v16u;
typedef __attribute__((ext_vector_type(8)))  unsigned short v8u;

__device__ __forceinline__ v16bf as_bf16x16(v16u x) {
    union { v16u u; v16bf b; } c; c.u = x; return c.b;
}

// float -> bf16 (round to nearest even)
__device__ __forceinline__ unsigned short f2bf(float f) {
    unsigned u = __float_as_uint(f);
    u += 0x7FFFu + ((u >> 16) & 1u);
    return (unsigned short)(u >> 16);
}

// ---------- WMMA fragment loaders (bf16, 16x16x32) ----------
// A matrix (16x32, MxK) from row-major memory with leading dim ld (elements).
// ISA layout: lane m = lane&15; elems 0..7 -> K = (lane>>4)*8 + e ; elems 8..15 -> K = 16+(lane>>4)*8+e
__device__ __forceinline__ v16bf load_a_frag(const unsigned short* base, int ld) {
    int lane = threadIdx.x & 31;
    const unsigned short* row = base + (lane & 15) * ld + ((lane >> 4) << 3);
    v8u lo = *(const v8u*)(row);
    v8u hi = *(const v8u*)(row + 16);
    v16u r;
#pragma unroll
    for (int i = 0; i < 8; ++i) { r[i] = lo[i]; r[i + 8] = hi[i]; }
    return as_bf16x16(r);
}

// B matrix (32x16, KxN) where memory holds T[n][k] (N rows, K contiguous), leading dim ldt.
// ISA layout: lane n = lane&15; elems 0..7 -> K = (lane>>4)*8 + e ; elems 8..15 -> K = 16+(lane>>4)*8+e
__device__ __forceinline__ v16bf load_b_fragT(const unsigned short* T, int ldt) {
    int lane = threadIdx.x & 31;
    const unsigned short* p = T + (lane & 15) * ldt + ((lane >> 4) << 3);
    v8u lo = *(const v8u*)(p);
    v8u hi = *(const v8u*)(p + 16);
    v16u r;
#pragma unroll
    for (int i = 0; i < 8; ++i) { r[i] = lo[i]; r[i + 8] = hi[i]; }
    return as_bf16x16(r);
}

__device__ __forceinline__ v8f wmma_bf16(v16bf a, v16bf b, v8f c) {
    return __builtin_amdgcn_wmma_f32_16x16x32_bf16(false, a, false, b, (short)0, c, false, false);
}

// gfx1250 async copy: global -> LDS, 16 bytes per lane, tracked by ASYNCcnt
__device__ __forceinline__ void async_g2l_b128(unsigned lds_off, const void* gaddr) {
    asm volatile("global_load_async_to_lds_b128 %0, %1, off"
                 :: "v"(lds_off), "v"(gaddr) : "memory");
}
__device__ __forceinline__ void wait_asynccnt0() {
    asm volatile("s_wait_asynccnt 0x0" ::: "memory");
}

// ---------- problem constants ----------
#define BATCH 8
#define CH    512
#define SLEN  1024
#define HEADS 8
#define HD    64
#define NGRP  32
#define QSCL  0.125f   // 64^-0.5

// ---------- K0: weights -> bf16 (fold q scale into qkv rows o<512) ----------
__global__ void k_cvt_w(const float* __restrict__ qkv_w, const float* __restrict__ proj_w,
                        unsigned short* __restrict__ wq, unsigned short* __restrict__ wp) {
    int i = blockIdx.x * 256 + threadIdx.x;
    const int NQ = 3 * CH * CH;
    if (i < NQ) {
        int o = i >> 9;
        float s = (o < CH) ? QSCL : 1.0f;
        wq[i] = f2bf(qkv_w[i] * s);
    }
    if (i < CH * CH) wp[i] = f2bf(proj_w[i]);
}

// ---------- K1: GroupNorm -> XnT [b][s][c] bf16 ----------
__global__ void k_gn(const float* __restrict__ X, const float* __restrict__ gw,
                     const float* __restrict__ gb, unsigned short* __restrict__ XnT) {
    int b = blockIdx.x >> 5;
    int g = blockIdx.x & 31;
    const float* xg = X + (size_t)b * CH * SLEN + (size_t)g * 16 * SLEN;

    float s = 0.f, ss = 0.f;
    for (int c = 0; c < 16; ++c)
        for (int i = threadIdx.x; i < SLEN; i += 256) {
            float v = xg[c * SLEN + i];
            s += v; ss += v * v;
        }
    __shared__ float rs[256], rss[256];
    rs[threadIdx.x] = s; rss[threadIdx.x] = ss;
    __syncthreads();
    for (int st = 128; st > 0; st >>= 1) {
        if (threadIdx.x < st) { rs[threadIdx.x] += rs[threadIdx.x + st]; rss[threadIdx.x] += rss[threadIdx.x + st]; }
        __syncthreads();
    }
    float mean = rs[0] * (1.0f / 16384.0f);
    float var  = rss[0] * (1.0f / 16384.0f) - mean * mean;
    float rstd = rsqrtf(var + 1e-5f);

    float wv[16], bv[16];
#pragma unroll
    for (int c = 0; c < 16; ++c) {
        wv[c] = gw[g * 16 + c] * rstd;
        bv[c] = gb[g * 16 + c] - mean * wv[c];
    }
    for (int sp = threadIdx.x; sp < SLEN; sp += 256) {
        alignas(16) unsigned short outv[16];
#pragma unroll
        for (int c = 0; c < 16; ++c) outv[c] = f2bf(xg[c * SLEN + sp] * wv[c] + bv[c]);
        unsigned short* dst = XnT + ((size_t)b * SLEN + sp) * CH + g * 16;
        *(v8u*)(dst)     = *(v8u*)(outv);
        *(v8u*)(dst + 8) = *(v8u*)(outv + 8);
    }
}

// ---------- K2: QKV GEMM, wave = 16(o) x 64(s) tile ----------
// All 8 waves of a block share one 16x512 weight tile -> async-stage it in LDS once.
// Inner loop: 2x-unrolled ping-pong, unconditional prefetch (1-step overfetch is padded/safe).
__global__ void k_qkv(const unsigned short* __restrict__ wq, const unsigned short* __restrict__ XnT,
                      const float* __restrict__ qkv_b,
                      unsigned short* __restrict__ qT, unsigned short* __restrict__ kT,
                      unsigned short* __restrict__ vmat) {
    __shared__ unsigned short Asm[16 * CH + 128];   // 16 KB weight tile + overfetch pad

    int wid = blockIdx.x * 8 + (threadIdx.x >> 5);
    int sG  = wid & 15;          // 16 groups of 64 s
    int tmp = wid >> 4;          // constant within block (8 waves never cross a 16-boundary)
    int oT  = tmp % 96;          // 96 o-tiles (1536/16)
    int b   = tmp / 96;
    int oBase = oT * 16, sBase = sG * 64;

    // ---- async stage A tile (rows oBase..oBase+15 of wq, contiguous) into LDS ----
    const unsigned short* Ag = wq + (size_t)oBase * CH;
    for (int i = threadIdx.x; i < (16 * CH) / 8; i += 256) {   // 1024 x 16B chunks
        unsigned lds_off = (unsigned)(unsigned long long)(const void*)&Asm[i * 8];
        async_g2l_b128(lds_off, (const void*)(Ag + i * 8));
    }
    wait_asynccnt0();
    __syncthreads();

    const unsigned short* Bt = XnT + ((size_t)b * SLEN + sBase) * CH;

    v8f acc[4] = {};
    v16bf a0 = load_a_frag(Asm, CH);
    v16bf b0[4];
#pragma unroll
    for (int t = 0; t < 4; ++t) b0[t] = load_b_fragT(Bt + (size_t)t * 16 * CH, CH);

    for (int kk = 0; kk < CH; kk += 64) {
        v16bf a1 = load_a_frag(Asm + kk + 32, CH);
        v16bf b1[4];
#pragma unroll
        for (int t = 0; t < 4; ++t) b1[t] = load_b_fragT(Bt + (size_t)t * 16 * CH + kk + 32, CH);
#pragma unroll
        for (int t = 0; t < 4; ++t) acc[t] = wmma_bf16(a0, b0[t], acc[t]);
        a0 = load_a_frag(Asm + kk + 64, CH);                 // tail overfetch -> pad
#pragma unroll
        for (int t = 0; t < 4; ++t) b0[t] = load_b_fragT(Bt + (size_t)t * 16 * CH + kk + 64, CH);
#pragma unroll
        for (int t = 0; t < 4; ++t) acc[t] = wmma_bf16(a1, b1[t], acc[t]);
    }

    int lane = threadIdx.x & 31;
    int nl = lane & 15;
    int part = oBase / CH;                 // 0=q 1=k 2=v (tile never straddles)
#pragma unroll
    for (int t = 0; t < 4; ++t) {
        int sidx = sBase + t * 16 + nl;
#pragma unroll
        for (int r = 0; r < 8; ++r) {
            int m = ((lane >> 4) << 3) + r;
            int o = oBase + m;
            float bias = qkv_b[o];
            if (part == 0) bias *= QSCL;
            unsigned short vb = f2bf(acc[t][r] + bias);
            int oin = o - part * CH;
            int h = oin >> 6, d = oin & 63;
            size_t bh = (size_t)b * HEADS + h;
            if (part == 0)      qT[(bh * SLEN + sidx) * HD + d] = vb;
            else if (part == 1) kT[(bh * SLEN + sidx) * HD + d] = vb;
            else                vmat[(bh * HD + d) * SLEN + sidx] = vb;
        }
    }
}

// ---------- K3: attention (block = one (b,h,16-query tile); 8 waves) ----------
#define PLD 1040   // P row stride (bf16), padded; 1040*2 bytes = 16B aligned
__global__ void k_attn(const unsigned short* __restrict__ qT, const unsigned short* __restrict__ kT,
                       const unsigned short* __restrict__ vmat, unsigned short* __restrict__ attnT) {
    int qTile = blockIdx.x & 63;
    int h     = (blockIdx.x >> 6) & 7;
    int b     = blockIdx.x >> 9;
    int w     = threadIdx.x >> 5;
    int lane  = threadIdx.x & 31;
    size_t bh = (size_t)b * HEADS + h;

    const unsigned short* Q = qT + (bh * SLEN + (size_t)qTile * 16) * HD;
    const unsigned short* K = kT + bh * SLEN * HD;
    const unsigned short* V = vmat + bh * HD * SLEN;

    __shared__ unsigned short P[16 * PLD + 64];   // + overfetch pad
    __shared__ float outAcc[16 * 64];
    __shared__ float wmax[8][16];
    __shared__ float wsum[8][16];
    __shared__ float rowmax[16];
    __shared__ float rowinv[16];

    for (int i = threadIdx.x; i < 16 * 64; i += 256) outAcc[i] = 0.f;

    // ---- phase 1: scores for key cols [w*128, w*128+128), pipelined (full unroll) ----
    v16bf aq0 = load_a_frag(Q + 0, HD);
    v16bf aq1 = load_a_frag(Q + 32, HD);
    int colBase = w * 128;
    v8f sc[8] = {};
    v16bf bk0 = load_b_fragT(K + (size_t)colBase * HD, HD);
    v16bf bk1 = load_b_fragT(K + (size_t)colBase * HD + 32, HD);
#pragma unroll
    for (int t = 0; t < 8; ++t) {
        // unconditional prefetch; t==7 overfetches into the adjacent ws buffer (unused)
        const unsigned short* Kt = K + (size_t)(colBase + (t + 1) * 16) * HD;
        v16bf nb0 = load_b_fragT(Kt + 0, HD);
        v16bf nb1 = load_b_fragT(Kt + 32, HD);
        sc[t] = wmma_bf16(aq0, bk0, sc[t]);
        sc[t] = wmma_bf16(aq1, bk1, sc[t]);
        bk0 = nb0; bk1 = nb1;
    }

    // ---- softmax: row max ----
    float mloc[8];
#pragma unroll
    for (int r = 0; r < 8; ++r) {
        float m = -3.0e38f;
#pragma unroll
        for (int t = 0; t < 8; ++t) m = fmaxf(m, sc[t][r]);
        mloc[r] = m;
    }
#pragma unroll
    for (int mask = 1; mask < 16; mask <<= 1)
#pragma unroll
        for (int r = 0; r < 8; ++r) mloc[r] = fmaxf(mloc[r], __shfl_xor(mloc[r], mask, 32));
    if ((lane & 15) == 0) {
        int mb = (lane >> 4) << 3;
#pragma unroll
        for (int r = 0; r < 8; ++r) wmax[w][mb + r] = mloc[r];
    }
    __syncthreads();
    if (threadIdx.x < 16) {
        float m = wmax[0][threadIdx.x];
#pragma unroll
        for (int ww = 1; ww < 8; ++ww) m = fmaxf(m, wmax[ww][threadIdx.x]);
        rowmax[threadIdx.x] = m;
    }
    __syncthreads();

    // ---- exp, row sum, write P (bf16) ----
    float sloc[8] = {0, 0, 0, 0, 0, 0, 0, 0};
#pragma unroll
    for (int t = 0; t < 8; ++t)
#pragma unroll
        for (int r = 0; r < 8; ++r) {
            int m = ((lane >> 4) << 3) + r;
            float p = __expf(sc[t][r] - rowmax[m]);
            sloc[r] += p;
            P[m * PLD + colBase + t * 16 + (lane & 15)] = f2bf(p);
        }
#pragma unroll
    for (int mask = 1; mask < 16; mask <<= 1)
#pragma unroll
        for (int r = 0; r < 8; ++r) sloc[r] += __shfl_xor(sloc[r], mask, 32);
    if ((lane & 15) == 0) {
        int mb = (lane >> 4) << 3;
#pragma unroll
        for (int r = 0; r < 8; ++r) wsum[w][mb + r] = sloc[r];
    }
    __syncthreads();
    if (threadIdx.x < 16) {
        float s = 0.f;
#pragma unroll
        for (int ww = 0; ww < 8; ++ww) s += wsum[ww][threadIdx.x];
        rowinv[threadIdx.x] = 1.0f / s;
    }
    __syncthreads();

    // ---- phase 2: out[16][64] = P[16][1024] x V^T ; wave w: d-tile (w&3), K half (w>>2) ----
    // 2x ping-pong, unconditional prefetch (overfetch padded/safe).
    int t4 = w & 3;
    int khalf = w >> 2;
    const unsigned short* Vt = V + (size_t)(t4 * 16) * SLEN;  // rows d, ld = SLEN
    v8f oa = {};
    int k0 = khalf * 512;
    v16bf av0 = load_a_frag(P + k0, PLD);
    v16bf bv0 = load_b_fragT(Vt + k0, SLEN);
    for (int kk = k0; kk < k0 + 512; kk += 64) {
        v16bf av1 = load_a_frag(P + kk + 32, PLD);
        v16bf bv1 = load_b_fragT(Vt + kk + 32, SLEN);
        oa = wmma_bf16(av0, bv0, oa);
        av0 = load_a_frag(P + kk + 64, PLD);        // tail overfetch -> pad
        bv0 = load_b_fragT(Vt + kk + 64, SLEN);     // tail overfetch -> adjacent buffer
        oa = wmma_bf16(av1, bv1, oa);
    }
#pragma unroll
    for (int r = 0; r < 8; ++r) {
        int m = ((lane >> 4) << 3) + r;
        int d = t4 * 16 + (lane & 15);
        atomicAdd(&outAcc[m * 64 + d], oa[r]);
    }
    __syncthreads();

    // ---- store to attnT [b][s][c] ----
    for (int i = threadIdx.x; i < 16 * 64; i += 256) {
        int m = i >> 6, d = i & 63;
        int sidx = qTile * 16 + m;
        attnT[((size_t)b * SLEN + sidx) * CH + h * HD + d] = f2bf(outAcc[i] * rowinv[m]);
    }
}

// ---------- K4: proj GEMM + bias + residual (fp32 out), 2x ping-pong ----------
__global__ void k_proj(const unsigned short* __restrict__ wp, const unsigned short* __restrict__ attnT,
                       const float* __restrict__ proj_b, const float* __restrict__ X,
                       float* __restrict__ out) {
    int wid = blockIdx.x * (blockDim.x >> 5) + (threadIdx.x >> 5);
    int sG  = wid & 15;
    int tmp = wid >> 4;
    int oT  = tmp & 31;
    int b   = tmp >> 5;
    int oBase = oT * 16, sBase = sG * 64;

    const unsigned short* A  = wp + (size_t)oBase * CH;
    const unsigned short* Bt = attnT + ((size_t)b * SLEN + sBase) * CH;

    v8f acc[4] = {};
    v16bf a0 = load_a_frag(A, CH);
    v16bf b0[4];
#pragma unroll
    for (int t = 0; t < 4; ++t) b0[t] = load_b_fragT(Bt + (size_t)t * 16 * CH, CH);

    for (int kk = 0; kk < CH; kk += 64) {
        v16bf a1 = load_a_frag(A + kk + 32, CH);
        v16bf b1[4];
#pragma unroll
        for (int t = 0; t < 4; ++t) b1[t] = load_b_fragT(Bt + (size_t)t * 16 * CH + kk + 32, CH);
#pragma unroll
        for (int t = 0; t < 4; ++t) acc[t] = wmma_bf16(a0, b0[t], acc[t]);
        a0 = load_a_frag(A + kk + 64, CH);           // tail overfetch -> next ws buffer
#pragma unroll
        for (int t = 0; t < 4; ++t) b0[t] = load_b_fragT(Bt + (size_t)t * 16 * CH + kk + 64, CH);
#pragma unroll
        for (int t = 0; t < 4; ++t) acc[t] = wmma_bf16(a1, b1[t], acc[t]);
    }

    int lane = threadIdx.x & 31;
#pragma unroll
    for (int t = 0; t < 4; ++t) {
        int sidx = sBase + t * 16 + (lane & 15);
#pragma unroll
        for (int r = 0; r < 8; ++r) {
            int m = ((lane >> 4) << 3) + r;
            int o = oBase + m;
            size_t idx = ((size_t)b * CH + o) * SLEN + sidx;
            out[idx] = acc[t][r] + proj_b[o] + X[idx];
        }
    }
}

// ---------- launcher ----------
extern "C" void kernel_launch(void* const* d_in, const int* in_sizes, int n_in,
                              void* d_out, int out_size, void* d_ws, size_t ws_size,
                              hipStream_t stream) {
    const float* X      = (const float*)d_in[0];
    const float* norm_w = (const float*)d_in[1];
    const float* norm_b = (const float*)d_in[2];
    const float* qkv_w  = (const float*)d_in[3];
    const float* qkv_b  = (const float*)d_in[4];
    const float* proj_w = (const float*)d_in[5];
    const float* proj_b = (const float*)d_in[6];
    float* out = (float*)d_out;

    unsigned short* ws = (unsigned short*)d_ws;
    size_t offXnT = 0;
    size_t offWq  = offXnT + (size_t)BATCH * SLEN * CH;     // 4M elems
    size_t offWp  = offWq  + (size_t)3 * CH * CH;           // +768K
    size_t offQT  = offWp  + (size_t)CH * CH;               // +256K
    size_t offKT  = offQT  + (size_t)BATCH * HEADS * SLEN * HD;
    size_t offV   = offKT  + (size_t)BATCH * HEADS * SLEN * HD;
    size_t offAT  = offV   + (size_t)BATCH * HEADS * HD * SLEN;
    // NOTE: ~1 KB of slack past offAT+4M elems is read (never written) by tail overfetch.

    unsigned short* XnT  = ws + offXnT;
    unsigned short* wq   = ws + offWq;
    unsigned short* wpb  = ws + offWp;
    unsigned short* qT   = ws + offQT;
    unsigned short* kT   = ws + offKT;
    unsigned short* vmat = ws + offV;
    unsigned short* aT   = ws + offAT;

    // K0: weight conversion (3*512*512 elems / 256)
    k_cvt_w<<<3072, 256, 0, stream>>>(qkv_w, proj_w, wq, wpb);
    // K1: group norm (B*NGRP blocks)
    k_gn<<<BATCH * NGRP, 256, 0, stream>>>(X, norm_w, norm_b, XnT);
    // K2: QKV GEMM — 12288 waves, 8 waves/block
    k_qkv<<<1536, 256, 0, stream>>>(wq, XnT, qkv_b, qT, kT, vmat);
    // K3: attention — B*H*(S/16) blocks
    k_attn<<<BATCH * HEADS * (SLEN / 16), 256, 0, stream>>>(qT, kT, vmat, aT);
    // K4: proj + residual — 4096 waves, 8 waves/block
    k_proj<<<512, 256, 0, stream>>>(wpb, aT, proj_b, X, out);
}
